// EdgePoint2Wrapper_4655744549323
// MI455X (gfx1250) — matrix operand order repaired and checked by reading, not compiled, not run on hardware
//
#include <hip/hip_runtime.h>

typedef _Float16 v16h __attribute__((ext_vector_type(16)));
typedef float    v8f  __attribute__((ext_vector_type(8)));

#define NB 4
#define TOPK 2048
#define CAND_MAX 131072

// ---------------- small device helpers ----------------
static __device__ __forceinline__ unsigned f2key(float f) {
  unsigned u = __float_as_uint(f);
  return (u & 0x80000000u) ? ~u : (u | 0x80000000u);
}
static __device__ __forceinline__ float key2f(unsigned k) {
  unsigned u = (k & 0x80000000u) ? (k ^ 0x80000000u) : ~k;
  return __uint_as_float(u);
}
static __device__ __forceinline__ float bilin_clamp(const _Float16* p, int H, int W,
                                                    float sy, float sx) {
  float y0f = floorf(sy), x0f = floorf(sx);
  float wy = sy - y0f, wx = sx - x0f;
  int y0 = (int)y0f, x0 = (int)x0f;
  int y1 = y0 + 1, x1 = x0 + 1;
  y0 = y0 < 0 ? 0 : (y0 > H - 1 ? H - 1 : y0);
  y1 = y1 < 0 ? 0 : (y1 > H - 1 ? H - 1 : y1);
  x0 = x0 < 0 ? 0 : (x0 > W - 1 ? W - 1 : x0);
  x1 = x1 < 0 ? 0 : (x1 > W - 1 ? W - 1 : x1);
  float v00 = (float)p[y0 * W + x0], v01 = (float)p[y0 * W + x1];
  float v10 = (float)p[y1 * W + x0], v11 = (float)p[y1 * W + x1];
  return v00 * (1.f - wx) * (1.f - wy) + v01 * wx * (1.f - wy) +
         v10 * (1.f - wx) * wy + v11 * wx * wy;
}

// ---------------- preprocessing ----------------
__global__ void k_zero_ctrl(unsigned* ctrl) {
  if (threadIdx.x < 32) ctrl[threadIdx.x] = 0u;
}

__global__ void k_gray_stats(const float* __restrict__ x, float* __restrict__ stats, int HW) {
  __shared__ float r1[256], r2[256];
  int b = blockIdx.y, tid = threadIdx.x;
  const float* p = x + (size_t)b * 3 * HW;
  float a1 = 0.f, a2 = 0.f;
  for (int i = blockIdx.x * 256 + tid; i < HW; i += gridDim.x * 256) {
    float g = (p[i] + p[i + HW] + p[i + 2 * HW]) * (1.f / 3.f);
    a1 += g; a2 += g * g;
  }
  r1[tid] = a1; r2[tid] = a2; __syncthreads();
  for (int s = 128; s > 0; s >>= 1) {
    if (tid < s) { r1[tid] += r1[tid + s]; r2[tid] += r2[tid + s]; }
    __syncthreads();
  }
  if (tid == 0) { atomicAdd(&stats[2 * b], r1[0]); atomicAdd(&stats[2 * b + 1], r2[0]); }
}

__global__ void k_norm(const float* __restrict__ x, const float* __restrict__ stats,
                       _Float16* __restrict__ gray, int HW) {
  int b = blockIdx.y;
  float mean = stats[2 * b] / (float)HW;
  float var = stats[2 * b + 1] / (float)HW - mean * mean;
  float rs = rsqrtf(var + 1e-5f);
  const float* p = x + (size_t)b * 3 * HW;
  for (int i = blockIdx.x * 256 + threadIdx.x; i < HW; i += gridDim.x * 256) {
    float g = (p[i] + p[i + HW] + p[i + 2 * HW]) * (1.f / 3.f);
    gray[(size_t)b * HW + i] = (_Float16)((g - mean) * rs);
  }
}

// stem: conv 1->8, k=4, stride 2, pad 1, BN+ReLU (tiny K, direct VALU conv)
__global__ void k_conv_b1a(const _Float16* __restrict__ gray, const float* __restrict__ w,
                           const float* __restrict__ scale, const float* __restrict__ bias,
                           _Float16* __restrict__ out) {
  const int HO = 512, WO = 512, HI = 1024, WI = 1024;
  int i = blockIdx.x * blockDim.x + threadIdx.x;
  if (i >= NB * HO * WO) return;
  int ox = i % WO; int t = i / WO; int oy = t % HO; int b = t / HO;
  float acc[8];
#pragma unroll
  for (int c = 0; c < 8; ++c) acc[c] = 0.f;
  const _Float16* gp = gray + (size_t)b * HI * WI;
#pragma unroll
  for (int kh = 0; kh < 4; ++kh) {
    int iy = 2 * oy - 1 + kh;
    if (iy < 0 || iy >= HI) continue;
#pragma unroll
    for (int kw = 0; kw < 4; ++kw) {
      int ix = 2 * ox - 1 + kw;
      if (ix < 0 || ix >= WI) continue;
      float v = (float)gp[iy * WI + ix];
#pragma unroll
      for (int c = 0; c < 8; ++c) acc[c] += v * w[c * 16 + kh * 4 + kw];
    }
  }
#pragma unroll
  for (int c = 0; c < 8; ++c) {
    float r = acc[c] * scale[c] + bias[c];
    r = r > 0.f ? r : 0.f;
    out[(((size_t)b * 8 + c) * HO + oy) * WO + ox] = (_Float16)r;
  }
}

// ---------------- weight packing into WMMA A-fragment layout ----------------
// layout: [OT][KT][lane 0..31][elem 0..15]; K-order: k = tap*ICP + ic, tap = kh*KW+kw
__global__ void k_pack_w(const float* __restrict__ w, _Float16* __restrict__ out,
                         int OT, int KT, int OC, int ICg, int ICP, int KH, int KW) {
  int i = blockIdx.x * blockDim.x + threadIdx.x;
  int total = OT * KT * 512;
  if (i >= total) return;
  int e = i & 15;
  int lane = (i >> 4) & 31;
  int kt = (i >> 9) % KT;
  int ot = (i >> 9) / KT;
  int m = lane & 15;
  int j = e >> 1, eo = e & 1;
  int kk = 2 * j + eo + ((lane >> 4) << 4);
  int k = kt * 32 + kk;
  int Ktot = KH * KW * ICP;
  float v = 0.f;
  int oc = ot * 16 + m;
  if (k < Ktot && oc < OC) {
    int tap = k / ICP; int ic = k - tap * ICP;
    if (ic < ICg) {
      int kh = tap / KW; int kw = tap - kh * KW;
      v = w[((oc * ICg + ic) * KH + kh) * KW + kw];
    }
  }
  out[i] = (_Float16)v;
}

// ---------------- implicit-GEMM WMMA convolution ----------------
// LDS tile is staged CHANNEL-LAST ([pixel][ic]): since k = tap*ICP + ic and ICP is a
// multiple of 16, each lane-half's 16 consecutive K values are 16 consecutive input
// channels at one (kh,kw) tap -> the whole B fragment is one aligned 32-byte LDS read.
// EPI bits: 1=BN(scale,bias)  2=bias-only  4=ReLU  8=residual-add(fp16)
template <int ICg, int ICP, int OCg, int KH, int KW, int PAD, int G, int EPI>
__global__ __launch_bounds__(128) void k_conv_wmma(
    const _Float16* __restrict__ X, const _Float16* __restrict__ Apk,
    const float* __restrict__ scale_, const float* __restrict__ bias_,
    const _Float16* __restrict__ resid, _Float16* __restrict__ Y,
    int H, int W, int CinT, int CoutT) {
  constexpr int MT = (OCg + 15) / 16;
  constexpr int Ktot = KH * KW * ICP;
  constexpr int KT = (Ktot + 31) / 32;
  constexpr int IH = 8 + KH - 1, IW = 8 + KW - 1;
  constexpr int NPIX = IH * IW;
  static_assert(ICP % 16 == 0, "ICP must be a multiple of 16");
  __shared__ __align__(32) _Float16 sx[NPIX * ICP];

  const int tid = threadIdx.x;
  const int lane = tid & 31, wave = tid >> 5;
  int z = blockIdx.z;
  const int mt = z % MT; z /= MT;
  const int g = z % G;  z /= G;
  const int b = z;
  const int ty0 = blockIdx.y * 8, tx0 = blockIdx.x * 8;

  // cooperative stage of input tile, channel-last, zero-padded
  for (int idx = tid; idx < ICP * NPIX; idx += 128) {
    int ic = idx / NPIX;
    int r = idx - ic * NPIX;
    int ly = r / IW, lx = r - ly * IW;
    int gy = ty0 + ly - PAD, gx = tx0 + lx - PAD;
    _Float16 v = (_Float16)0.f;
    if (ic < ICg && gy >= 0 && gy < H && gx >= 0 && gx < W)
      v = X[(((size_t)b * CinT + g * ICg + ic) * H + gy) * W + gx];
    sx[r * ICP + ic] = v;
  }
  __syncthreads();

  const v16h* ap = (const v16h*)Apk + ((size_t)(g * MT + mt) * KT * 32 + lane);
  v8f acc = {};
  const int n = lane & 15;
  const int py = 2 * wave + (n >> 3), px = n & 7;
  const int kupper = (lane >> 4) << 4;  // B-operand: lanes 16..31 hold K+16

  for (int kt = 0; kt < KT; ++kt) {
    v16h af = ap[(size_t)kt * 32];
    if (kt + 1 < KT) __builtin_prefetch((const void*)(ap + (size_t)(kt + 1) * 32), 0, 0);
    const int k0 = kt * 32 + kupper;        // 16-aligned; never crosses a tap boundary
    v16h bf = {};
    if (k0 < Ktot) {
      int tap = k0 / ICP;
      int icb = k0 - tap * ICP;
      int kh = tap / KW, kw = tap - kh * KW;
      bf = *(const v16h*)&sx[((py + kh) * IW + (px + kw)) * ICP + icb];
    }
    acc = __builtin_amdgcn_wmma_f32_16x16x32_f16(false, af, false, bf, (short)0, acc,
                                                 false, false);
  }

  const int oy = ty0 + py, ox = tx0 + px;
  if (oy >= H || ox >= W) return;
  const int mbase = (lane >> 4) << 3;
#pragma unroll
  for (int j = 0; j < 8; ++j) {
    int oc = mt * 16 + mbase + j;
    if (oc >= OCg) continue;
    int ocg = g * OCg + oc;
    float val = acc[j];
    if (EPI & 1) val = val * scale_[ocg] + bias_[ocg];
    if (EPI & 2) val += bias_[ocg];
    if (EPI & 8) val += (float)resid[(((size_t)b * CoutT + ocg) * H + oy) * W + ox];
    if (EPI & 4) val = val > 0.f ? val : 0.f;
    Y[(((size_t)b * CoutT + ocg) * H + oy) * W + ox] = (_Float16)val;
  }
}

// ---------------- pooling / resize / concat helpers ----------------
__global__ void k_avgpool(const _Float16* __restrict__ in, _Float16* __restrict__ out,
                          int C, int Hi, int Wi, int k, int CoutT, int cofs) {
  int Ho = Hi / k, Wo = Wi / k;
  int i = blockIdx.x * 256 + threadIdx.x;
  int total = NB * C * Ho * Wo;
  if (i >= total) return;
  int x = i % Wo; int t = i / Wo; int y = t % Ho; t /= Ho; int c = t % C; int b = t / C;
  float s = 0.f;
  for (int dy = 0; dy < k; ++dy)
    for (int dx = 0; dx < k; ++dx)
      s += (float)in[(((size_t)b * C + c) * Hi + y * k + dy) * Wi + x * k + dx];
  out[(((size_t)b * CoutT + cofs + c) * Ho + y) * Wo + x] = (_Float16)(s / (float)(k * k));
}

__global__ void k_up_cat(const _Float16* __restrict__ in, _Float16* __restrict__ out,
                         int C, int Hi, int Wi, int s, int CoutT, int cofs) {
  int Ho = Hi * s, Wo = Wi * s;
  int i = blockIdx.x * 256 + threadIdx.x;
  int total = NB * C * Ho * Wo;
  if (i >= total) return;
  int x = i % Wo; int t = i / Wo; int y = t % Ho; t /= Ho; int c = t % C; int b = t / C;
  float sy = (y + 0.5f) / (float)s - 0.5f;
  float sx = (x + 0.5f) / (float)s - 0.5f;
  float v = bilin_clamp(in + ((size_t)b * C + c) * Hi * Wi, Hi, Wi, sy, sx);
  out[(((size_t)b * CoutT + cofs + c) * Ho + y) * Wo + x] = (_Float16)v;
}

// s[B,8,512,512] += up4(c2o[B,8,128,128]) + up16(c3o[B,8,32,32])
__global__ void k_score_add(_Float16* __restrict__ s, const _Float16* __restrict__ c2o,
                            const _Float16* __restrict__ c3o) {
  int i = blockIdx.x * 256 + threadIdx.x;
  int total = NB * 8 * 512 * 512;
  if (i >= total) return;
  int x = i % 512; int t = i / 512; int y = t % 512; t /= 512; int c = t % 8; int b = t / 8;
  float add =
      bilin_clamp(c2o + ((size_t)b * 8 + c) * 128 * 128, 128, 128,
                  (y + 0.5f) / 4.f - 0.5f, (x + 0.5f) / 4.f - 0.5f) +
      bilin_clamp(c3o + ((size_t)b * 8 + c) * 32 * 32, 32, 32,
                  (y + 0.5f) / 16.f - 0.5f, (x + 0.5f) / 16.f - 0.5f);
  size_t o = (((size_t)b * 8 + c) * 512 + y) * 512 + x;
  s[o] = (_Float16)((float)s[o] + add);
}

// pixel_shuffle r=2: [B,4,512,512] fp16 -> [B,1024,1024] fp32
__global__ void k_pixshuf(const _Float16* __restrict__ in, float* __restrict__ out) {
  int i = blockIdx.x * 256 + threadIdx.x;
  int total = NB * 1024 * 1024;
  if (i >= total) return;
  int x = i % 1024; int t = i / 1024; int y = t % 1024; int b = t / 1024;
  int ch = (y & 1) * 2 + (x & 1);
  out[(size_t)b * 1048576 + (size_t)y * 1024 + x] =
      (float)in[(((size_t)b * 4 + ch) * 512 + (y >> 1)) * 512 + (x >> 1)];
}

// ---------------- detection: NMS + candidates + exact top-k ----------------
__global__ void k_detect(const float* __restrict__ det, unsigned long long* __restrict__ cands,
                         int* __restrict__ candcnt) {
  const int H = 1024, W = 1024;
  int i = blockIdx.x * 256 + threadIdx.x;
  if (i >= NB * H * W) return;
  int x = i % W; int t = i / W; int y = t % H; int b = t / H;
  if (x < 4 || x >= W - 4 || y < 4 || y >= H - 4) return;
  const float* p = det + (size_t)b * H * W;
  float v = p[y * W + x];
  if (!(v > -5.0f)) return;
  float m = -INFINITY;
#pragma unroll
  for (int dy = -2; dy <= 2; ++dy)
#pragma unroll
    for (int dx = -2; dx <= 2; ++dx) {
      float q = p[(y + dy) * W + (x + dx)];
      m = q > m ? q : m;
    }
  if (v >= m) {
    int pos = atomicAdd(&candcnt[b], 1);
    if (pos < CAND_MAX) {
      unsigned idx = (unsigned)(y * W + x);
      cands[(size_t)b * CAND_MAX + pos] =
          ((unsigned long long)f2key(v) << 32) | (unsigned)(0xFFFFFFFFu - idx);
    }
  }
}

// 4-pass byte radix-select of the exact key threshold for top TOPK (per image)
__global__ void k_select(const unsigned long long* __restrict__ cands,
                         const int* __restrict__ candcnt, unsigned* __restrict__ thresh,
                         unsigned* __restrict__ take_eq) {
  int b = blockIdx.x, tid = threadIdx.x;
  int n = candcnt[b]; n = n < CAND_MAX ? n : CAND_MAX;
  __shared__ unsigned hist[256];
  __shared__ unsigned sprefix, sneed;
  if (n <= TOPK) {
    if (tid == 0) { thresh[b] = 0u; take_eq[b] = 0u; }
    return;
  }
  if (tid == 0) { sprefix = 0u; sneed = TOPK; }
  __syncthreads();
  for (int shift = 24; shift >= 0; shift -= 8) {
    hist[tid] = 0u;
    __syncthreads();
    unsigned pref = sprefix;
    unsigned himask = (shift == 24) ? 0u : (0xFFFFFFFFu << (shift + 8));
    for (int i = tid; i < n; i += 256) {
      unsigned k = (unsigned)(cands[(size_t)b * CAND_MAX + i] >> 32);
      if ((k & himask) == pref) atomicAdd(&hist[(k >> shift) & 255], 1u);
    }
    __syncthreads();
    if (tid == 0) {
      unsigned need = sneed, cum = 0u;
      int v = 255;
      for (; v > 0; --v) {
        if (cum + hist[v] >= need) break;
        cum += hist[v];
      }
      sprefix = pref | ((unsigned)v << shift);
      sneed = need - cum;
    }
    __syncthreads();
  }
  if (tid == 0) { thresh[b] = sprefix; take_eq[b] = sneed; }
}

#define SENTVAL ((((unsigned long long)0x007FFFFFull) << 32) | 0xFFFFFFFFull)

__global__ void k_fill(unsigned long long* sel) {
  int i = blockIdx.x * 256 + threadIdx.x;
  if (i < NB * TOPK) sel[i] = SENTVAL;
}

__global__ void k_compact(const unsigned long long* __restrict__ cands,
                          const int* __restrict__ candcnt, const unsigned* __restrict__ thresh,
                          const unsigned* __restrict__ take_eq,
                          unsigned long long* __restrict__ sel, int* __restrict__ selcnt,
                          int* __restrict__ eqcnt) {
  int b = blockIdx.y;
  int n = candcnt[b]; n = n < CAND_MAX ? n : CAND_MAX;
  unsigned thr = thresh[b];
  for (int i = blockIdx.x * 256 + threadIdx.x; i < n; i += gridDim.x * 256) {
    unsigned long long c = cands[(size_t)b * CAND_MAX + i];
    unsigned k = (unsigned)(c >> 32);
    if (k > thr) {
      int p = atomicAdd(&selcnt[b], 1);
      if (p < TOPK) sel[(size_t)b * TOPK + p] = c;
    } else if (k == thr && thr != 0u) {
      int e = atomicAdd(&eqcnt[b], 1);
      if (e < (int)take_eq[b]) {
        int p = atomicAdd(&selcnt[b], 1);
        if (p < TOPK) sel[(size_t)b * TOPK + p] = c;
      }
    }
  }
}

// bitonic sort 2048 (descending; composite key breaks ties toward lower index)
__global__ __launch_bounds__(1024) void k_sort(unsigned long long* __restrict__ sel,
                                               float* __restrict__ out_kpts,
                                               float* __restrict__ out_scores) {
  __shared__ unsigned long long s[TOPK];
  int b = blockIdx.x, tid = threadIdx.x;
  for (int i = tid; i < TOPK; i += 1024) s[i] = sel[(size_t)b * TOPK + i];
  __syncthreads();
  for (unsigned k = 2; k <= TOPK; k <<= 1) {
    for (unsigned j = k >> 1; j > 0; j >>= 1) {
      for (unsigned i = tid; i < TOPK; i += 1024) {
        unsigned ixj = i ^ j;
        if (ixj > i) {
          bool asc = (i & k) != 0;  // overall descending
          unsigned long long a = s[i], c = s[ixj];
          if (asc ? (a > c) : (a < c)) { s[i] = c; s[ixj] = a; }
        }
      }
      __syncthreads();
    }
  }
  for (int i = tid; i < TOPK; i += 1024) {
    unsigned long long c = s[i];
    unsigned key = (unsigned)(c >> 32);
    unsigned idx = 0xFFFFFFFFu - (unsigned)c;
    float kx = (float)(idx % 1024u), ky = (float)(idx / 1024u);
    out_kpts[((size_t)b * TOPK + i) * 2 + 0] = kx;
    out_kpts[((size_t)b * TOPK + i) * 2 + 1] = ky;
    out_scores[(size_t)b * TOPK + i] = key2f(key);
  }
}

// one wave per keypoint: lane = descriptor channel, bilinear zero-pad + L2 norm
__global__ void k_sample(const _Float16* __restrict__ desc, const float* __restrict__ kpts,
                         float* __restrict__ outd) {
  int gw = (blockIdx.x * blockDim.x + threadIdx.x) >> 5;
  int lane = threadIdx.x & 31;
  if (gw >= NB * TOPK) return;
  int b = gw / TOPK, kp = gw - b * TOPK;
  float kx = kpts[((size_t)b * TOPK + kp) * 2 + 0];
  float ky = kpts[((size_t)b * TOPK + kp) * 2 + 1];
  float px = (kx + 0.5f) * 0.25f - 0.5f;
  float py = (ky + 0.5f) * 0.25f - 0.5f;
  float x0f = floorf(px), y0f = floorf(py);
  float wx = px - x0f, wy = py - y0f;
  int x0 = (int)x0f, y0 = (int)y0f;
  const _Float16* p = desc + ((size_t)b * 32 + lane) * 65536;
  auto g = [&](int xi, int yi) -> float {
    if (xi < 0 || xi >= 256 || yi < 0 || yi >= 256) return 0.f;
    return (float)p[yi * 256 + xi];
  };
  float v = g(x0, y0) * (1.f - wx) * (1.f - wy) + g(x0 + 1, y0) * wx * (1.f - wy) +
            g(x0, y0 + 1) * (1.f - wx) * wy + g(x0 + 1, y0 + 1) * wx * wy;
  float ss = v * v;
  for (int o = 16; o > 0; o >>= 1) ss += __shfl_xor(ss, o, 32);
  float nrm = sqrtf(ss);
  nrm = nrm > 1e-12f ? nrm : 1e-12f;
  outd[((size_t)b * TOPK + kp) * 32 + lane] = v / nrm;
}

// ---------------- host-side launch plumbing ----------------
template <int ICg, int ICP, int OCg, int KH, int KW, int PAD, int G, int EPI>
static void conv_launch(const _Float16* X, const _Float16* Apk, const float* sc,
                        const float* bi, const _Float16* res, _Float16* Y, int H, int W,
                        int CinT, int CoutT, hipStream_t stream) {
  constexpr int MT = (OCg + 15) / 16;
  dim3 grid((W + 7) / 8, (H + 7) / 8, NB * G * MT);
  k_conv_wmma<ICg, ICP, OCg, KH, KW, PAD, G, EPI>
      <<<grid, 128, 0, stream>>>(X, Apk, sc, bi, res, Y, H, W, CinT, CoutT);
}

struct PK { int widx, OT, KT, OC, ICg, ICP, KH, KW; };

extern "C" void kernel_launch(void* const* d_in, const int* in_sizes, int n_in,
                              void* d_out, int out_size, void* d_ws, size_t ws_size,
                              hipStream_t stream) {
  (void)in_sizes; (void)n_in; (void)out_size;
  char* ws = (char*)d_ws;
  size_t o = 0;
  auto alloc = [&](size_t bytes) -> size_t {
    size_t r = (o + 255) & ~(size_t)255;
    o = r + bytes;
    return r;
  };

  const size_t o_gray = alloc(8388608);      // [4,1024,1024] fp16 (later: sh3 out [4,4,512,512])
  const size_t o_b1a  = alloc(16777216);     // [4,8,512,512] fp16 (later: ssum [4,8,512,512])
  const size_t o_b1b  = alloc(33554432);     // [4,16,512,512] fp16 (later: sh2 out)
  const size_t o_t1   = alloc(33554432);     // b1c.l1 out (later: sh1 out)
  const size_t o_x1   = alloc(33554432);
  const size_t o_x2q  = alloc(8388608);      // _x2 [4,16,256,256]
  const size_t o_p128 = alloc(2097152);
  const size_t o_b2l1 = alloc(4194304);
  const size_t o_b2sc = alloc(4194304);
  const size_t o_x2   = alloc(4194304);
  const size_t o_p32  = alloc(262144);
  const size_t o_b3l1 = alloc(393216);
  const size_t o_b3sc = alloc(393216);
  const size_t o_x3   = alloc(393216);
  const size_t o_dcat = alloc(50331648);     // [4,96,256,256] fp16 (later: dh2 out, then cands)
  const size_t o_dh1  = alloc(50331648);     // dh1 out (later: raw_detect fp32)
  const size_t o_rdsc = alloc(16777216);     // raw_desc [4,32,256,256] fp16
  const size_t o_c2o  = alloc(1048576);
  const size_t o_c3o  = alloc(65536);
  const size_t o_sel  = alloc((size_t)NB * TOPK * 8);
  const size_t o_ctrl = alloc(256);
  const size_t o_pkw  = alloc(2097152);      // packed weights
  if (ws_size < o) return;

  _Float16* gray  = (_Float16*)(ws + o_gray);
  _Float16* yb1a  = (_Float16*)(ws + o_b1a);
  _Float16* yb1b  = (_Float16*)(ws + o_b1b);
  _Float16* t1    = (_Float16*)(ws + o_t1);
  _Float16* x1    = (_Float16*)(ws + o_x1);
  _Float16* x2q   = (_Float16*)(ws + o_x2q);
  _Float16* p128  = (_Float16*)(ws + o_p128);
  _Float16* b2l1  = (_Float16*)(ws + o_b2l1);
  _Float16* b2sc  = (_Float16*)(ws + o_b2sc);
  _Float16* x2    = (_Float16*)(ws + o_x2);
  _Float16* p32   = (_Float16*)(ws + o_p32);
  _Float16* b3l1  = (_Float16*)(ws + o_b3l1);
  _Float16* b3sc  = (_Float16*)(ws + o_b3sc);
  _Float16* x3    = (_Float16*)(ws + o_x3);
  _Float16* dcat  = (_Float16*)(ws + o_dcat);
  _Float16* dh1   = (_Float16*)(ws + o_dh1);
  _Float16* dh2   = dcat;                      // slab reuse
  _Float16* rdesc = (_Float16*)(ws + o_rdsc);
  _Float16* c2o   = (_Float16*)(ws + o_c2o);
  _Float16* c3o   = (_Float16*)(ws + o_c3o);
  _Float16* ssum  = yb1a;                      // slab reuse
  _Float16* sh1o  = t1;
  _Float16* sh2o  = yb1b;
  _Float16* sh3o  = gray;
  float*    rdet  = (float*)(ws + o_dh1);      // slab reuse
  unsigned long long* cands = (unsigned long long*)(ws + o_dcat);  // slab reuse
  unsigned long long* sel   = (unsigned long long*)(ws + o_sel);
  unsigned* ctrl  = (unsigned*)(ws + o_ctrl);
  float*    stats   = (float*)ctrl;            // [8]
  int*      candcnt = (int*)(ctrl + 8);        // [4]
  int*      selcnt  = (int*)(ctrl + 12);       // [4]
  int*      eqcnt   = (int*)(ctrl + 16);       // [4]
  unsigned* thresh  = ctrl + 20;               // [4]
  unsigned* takeq   = ctrl + 24;               // [4]
  _Float16* packed  = (_Float16*)(ws + o_pkw);

  float* out_kpts   = (float*)d_out;
  float* out_scores = out_kpts + (size_t)NB * TOPK * 2;
  float* out_desc   = out_scores + (size_t)NB * TOPK;

  auto F = [&](int i) { return (const float*)d_in[i]; };

  // ---- packed weight table (insertion-order flattening of setup_inputs) ----
  static const PK pk[18] = {
      {4, 1, 5, 16, 8, 16, 3, 3},    // b1b
      {7, 1, 5, 16, 16, 16, 3, 3},   // b1c.l1
      {10, 1, 5, 16, 16, 16, 3, 3},  // b1c.l2
      {13, 2, 5, 32, 16, 16, 3, 3},  // b2.l1
      {16, 2, 9, 32, 32, 32, 3, 3},  // b2.l2
      {19, 2, 1, 32, 16, 16, 1, 1},  // b2.sc
      {22, 3, 9, 48, 32, 32, 3, 3},  // b3.l1
      {25, 3, 14, 48, 48, 48, 3, 3}, // b3.l2
      {28, 3, 1, 48, 32, 32, 1, 1},  // b3.sc
      {31, 6, 3, 96, 96, 96, 1, 1},  // dh1
      {33, 6, 5, 96, 16, 16, 3, 3},  // dh2 (grouped, 6x 16->16)
      {36, 2, 3, 32, 96, 96, 1, 1},  // dh3
      {38, 1, 1, 8, 16, 16, 1, 1},   // c1
      {40, 1, 1, 8, 32, 32, 1, 1},   // c2
      {42, 1, 2, 8, 48, 48, 1, 1},   // c3
      {44, 1, 5, 8, 8, 16, 3, 3},    // sh1
      {46, 1, 5, 8, 8, 16, 3, 3},    // sh2
      {48, 1, 5, 4, 8, 16, 3, 3},    // sh3
  };
  size_t pofs[18];
  {
    size_t cur = 0;
    for (int i = 0; i < 18; ++i) {
      pofs[i] = cur;
      cur += (size_t)pk[i].OT * pk[i].KT * 512;
    }
  }
  auto P = [&](int i) { return (const _Float16*)(packed + pofs[i]); };

  // ---- launches ----
  k_zero_ctrl<<<1, 64, 0, stream>>>(ctrl);
  k_gray_stats<<<dim3(256, NB), 256, 0, stream>>>(F(0), stats, 1024 * 1024);
  k_norm<<<dim3(2048, NB), 256, 0, stream>>>(F(0), stats, gray, 1024 * 1024);
  k_conv_b1a<<<(NB * 512 * 512 + 255) / 256, 256, 0, stream>>>(gray, F(1), F(2), F(3), yb1a);

  for (int i = 0; i < 18; ++i) {
    int tot = pk[i].OT * pk[i].KT * 512;
    k_pack_w<<<(tot + 255) / 256, 256, 0, stream>>>(
        F(pk[i].widx), packed + pofs[i], pk[i].OT, pk[i].KT, pk[i].OC, pk[i].ICg,
        pk[i].ICP, pk[i].KH, pk[i].KW);
  }

  // backbone
  conv_launch<8, 16, 16, 3, 3, 1, 1, 5>(yb1a, P(0), F(5), F(6), nullptr, yb1b, 512, 512, 8, 16, stream);
  conv_launch<16, 16, 16, 3, 3, 1, 1, 5>(yb1b, P(1), F(8), F(9), nullptr, t1, 512, 512, 16, 16, stream);
  conv_launch<16, 16, 16, 3, 3, 1, 1, 13>(t1, P(2), F(11), F(12), yb1b, x1, 512, 512, 16, 16, stream);

  k_avgpool<<<(NB * 16 * 256 * 256 + 255) / 256, 256, 0, stream>>>(x1, x2q, 16, 512, 512, 2, 16, 0);
  k_avgpool<<<(NB * 16 * 256 * 256 + 255) / 256, 256, 0, stream>>>(x2q, dcat, 16, 256, 256, 1, 96, 0);
  k_avgpool<<<(NB * 16 * 128 * 128 + 255) / 256, 256, 0, stream>>>(x2q, p128, 16, 256, 256, 2, 16, 0);

  conv_launch<16, 16, 32, 1, 1, 0, 1, 1>(p128, P(5), F(20), F(21), nullptr, b2sc, 128, 128, 16, 32, stream);
  conv_launch<16, 16, 32, 3, 3, 1, 1, 5>(p128, P(3), F(14), F(15), nullptr, b2l1, 128, 128, 16, 32, stream);
  conv_launch<32, 32, 32, 3, 3, 1, 1, 13>(b2l1, P(4), F(17), F(18), b2sc, x2, 128, 128, 32, 32, stream);

  k_avgpool<<<(NB * 32 * 32 * 32 + 255) / 256, 256, 0, stream>>>(x2, p32, 32, 128, 128, 4, 32, 0);

  conv_launch<32, 32, 48, 1, 1, 0, 1, 1>(p32, P(8), F(29), F(30), nullptr, b3sc, 32, 32, 32, 48, stream);
  conv_launch<32, 32, 48, 3, 3, 1, 1, 5>(p32, P(6), F(23), F(24), nullptr, b3l1, 32, 32, 32, 48, stream);
  conv_launch<48, 48, 48, 3, 3, 1, 1, 13>(b3l1, P(7), F(26), F(27), b3sc, x3, 32, 32, 48, 48, stream);

  // descriptor head: concat + dh1 + dh2(grouped) + dh3
  k_up_cat<<<(NB * 32 * 256 * 256 + 255) / 256, 256, 0, stream>>>(x2, dcat, 32, 128, 128, 2, 96, 16);
  k_up_cat<<<(NB * 48 * 256 * 256 + 255) / 256, 256, 0, stream>>>(x3, dcat, 48, 32, 32, 8, 96, 48);
  conv_launch<96, 96, 96, 1, 1, 0, 1, 2>(dcat, P(9), nullptr, F(32), nullptr, dh1, 256, 256, 96, 96, stream);
  conv_launch<16, 16, 16, 3, 3, 1, 6, 5>(dh1, P(10), F(34), F(35), nullptr, dh2, 256, 256, 96, 96, stream);
  conv_launch<96, 96, 32, 1, 1, 0, 1, 2>(dh2, P(11), nullptr, F(37), nullptr, rdesc, 256, 256, 96, 32, stream);

  // score head
  conv_launch<16, 16, 8, 1, 1, 0, 1, 2>(x1, P(12), nullptr, F(39), nullptr, ssum, 512, 512, 16, 8, stream);
  conv_launch<32, 32, 8, 1, 1, 0, 1, 2>(x2, P(13), nullptr, F(41), nullptr, c2o, 128, 128, 32, 8, stream);
  conv_launch<48, 48, 8, 1, 1, 0, 1, 2>(x3, P(14), nullptr, F(43), nullptr, c3o, 32, 32, 48, 8, stream);
  k_score_add<<<(NB * 8 * 512 * 512 + 255) / 256, 256, 0, stream>>>(ssum, c2o, c3o);
  conv_launch<8, 16, 8, 3, 3, 1, 1, 6>(ssum, P(15), nullptr, F(45), nullptr, sh1o, 512, 512, 8, 8, stream);
  conv_launch<8, 16, 8, 3, 3, 1, 1, 6>(sh1o, P(16), nullptr, F(47), nullptr, sh2o, 512, 512, 8, 8, stream);
  conv_launch<8, 16, 4, 3, 3, 1, 1, 2>(sh2o, P(17), nullptr, F(49), nullptr, sh3o, 512, 512, 8, 4, stream);
  k_pixshuf<<<(NB * 1024 * 1024 + 255) / 256, 256, 0, stream>>>(sh3o, rdet);

  // detection + exact top-k + sort
  k_detect<<<(NB * 1024 * 1024 + 255) / 256, 256, 0, stream>>>(rdet, cands, candcnt);
  k_select<<<NB, 256, 0, stream>>>(cands, candcnt, thresh, takeq);
  k_fill<<<(NB * TOPK + 255) / 256, 256, 0, stream>>>(sel);
  k_compact<<<dim3(512, NB), 256, 0, stream>>>(cands, candcnt, thresh, takeq, sel, selcnt, eqcnt);
  k_sort<<<NB, 1024, 0, stream>>>(sel, out_kpts, out_scores);

  // descriptor sampling + L2 norm
  k_sample<<<(NB * TOPK * 32 + 255) / 256, 256, 0, stream>>>(rdesc, out_kpts, out_desc);
}
